// IMLayer_7550552506848
// MI455X (gfx1250) — matrix operand order repaired and checked by reading, not compiled, not run on hardware
//
#include <hip/hip_runtime.h>
#include <math.h>

#define BATCHN 8
#define SEQLEN 8192
#define HD 128
#define PD 128
#define PTD 256          // P*2 (p,t) packed
#define CHUNK 32
#define NCH (SEQLEN / CHUNK)

// LDS row strides (elements): row offsets land on distinct banks (stride %64 dwords == 4)
#define XH_S 136   // f16: 272B/row, 16B aligned
#define XF_S 132   // f32: 132 dwords/row
#define BU_S 264   // f16: 528B/row, 16B aligned

typedef _Float16 h8   __attribute__((ext_vector_type(8)));
typedef _Float16 v16h __attribute__((ext_vector_type(16)));
typedef float    v8f  __attribute__((ext_vector_type(8)));

// ---- ws layout (in floats) ----
#define OFF_CF 0ull                                       // chunk finals: B*NCH*PTD*2 f32
#define OFF_CA (OFF_CF + (size_t)BATCHN * NCH * PTD * 2)  // carries
#define OFF_BM (OFF_CA + (size_t)BATCHN * NCH * PTD * 2)  // BmatT[pt][h] f16
#define OFF_CM (OFF_BM + (size_t)PTD * HD / 2)            // CmatT[h][k]  f16
#define OFF_PR (OFF_CM + (size_t)HD * PTD / 2)            // params 10*PD f32

// params: [0]=M11 [1]=M12 [2]=M21 [3]=M22 [4]=f1s [5]=f2s [6..9]=M^CHUNK
#define PRM(k, p) prm[(k) * PD + (p)]

#define WMMA_F16(A, B, C) \
    __builtin_amdgcn_wmma_f32_16x16x32_f16(false, (A), false, (B), (short)0, (C), false, false)

__global__ __launch_bounds__(128)
void k_params(const float* __restrict__ stepsp, const float* __restrict__ Adiag,
              const float* __restrict__ Bin, const float* __restrict__ Cin,
              _Float16* __restrict__ BmatT, _Float16* __restrict__ CmatT,
              float* __restrict__ prm) {
    int p = threadIdx.x;  // 0..127 (also reused as h below)
    float st = 1.0f / (1.0f + expf(-stepsp[p]));
    float A  = fmaxf(Adiag[p], 0.0f);
    float s2A   = st * st * A;
    float schur = 1.0f / (1.0f + s2A);
    float M11 = 1.0f - s2A * schur;
    float M12 = -st * A * schur;
    float M21 = st * schur;
    float M22 = schur;
    PRM(0, p) = M11; PRM(1, p) = M12; PRM(2, p) = M21; PRM(3, p) = M22;
    PRM(4, p) = M11 * st;
    PRM(5, p) = M21 * st;
    float a = 1.f, b = 0.f, c = 0.f, d = 1.f;
    for (int i = 0; i < CHUNK; ++i) {
        float na = M11 * a + M12 * c, nb = M11 * b + M12 * d;
        float nc = M21 * a + M22 * c, nd = M21 * b + M22 * d;
        a = na; b = nb; c = nc; d = nd;
    }
    PRM(6, p) = a; PRM(7, p) = b; PRM(8, p) = c; PRM(9, p) = d;

    int h = threadIdx.x;
    for (int pp = 0; pp < PD; ++pp) {
        for (int t = 0; t < 2; ++t) {
            int k = pp * 2 + t;
            BmatT[k * HD + h] = (_Float16)Bin[((size_t)pp * HD + h) * 2 + t];
            float cv = Cin[((size_t)h * PD + pp) * 2 + t];
            CmatT[h * PTD + k] = (_Float16)(t ? -cv : cv);
        }
    }
}

// A-fragment for V_WMMA_F32_16X16X32_F16 from a row-major f16 LDS row.
// halves 0..15 = K(k0+2*khalf+0..15) (contiguous, matches B-fragment order)
__device__ __forceinline__ v16h load_afrag(const _Float16* rowp, int k0, int khalf) {
    h8 lo = *(const h8*)(rowp + k0 + 2 * khalf);
    h8 hi = *(const h8*)(rowp + k0 + 2 * khalf + 8);
    return __builtin_shufflevector(lo, hi, 0, 1, 2, 3, 4, 5, 6, 7,
                                           8, 9, 10, 11, 12, 13, 14, 15);
}

// GEMM1 tile: buh[32][BU_S] (f16) = xh[32][XH_S] (f16) @ BmatT (K=HD, N=PTD).
// Register-blocked 2x2 per wave (32 l x 32 pt): 4 WMMAs per k-step off 4 fragment loads.
__device__ __forceinline__ void gemm1_tile(const _Float16* xh, const _Float16* __restrict__ BmatT,
                                           _Float16* buh, int tid) {
    int wave = tid >> 5, lane = tid & 31;
    int lrow  = lane & 15;
    int khalf = (lane >> 4) * 8;
    int bkoff = (lane >> 4) * 16;
    int n0 = wave * 32;  // this wave covers pt columns [n0, n0+32)
    v8f acc00 = {}, acc01 = {}, acc10 = {}, acc11 = {};
#pragma unroll
    for (int k0 = 0; k0 < HD; k0 += 32) {
        v16h a0 = load_afrag(xh + lrow * XH_S, k0, khalf);
        v16h a1 = load_afrag(xh + (16 + lrow) * XH_S, k0, khalf);
        v16h b0 = *(const v16h*)(BmatT + (n0 + lrow) * HD + k0 + bkoff);
        v16h b1 = *(const v16h*)(BmatT + (n0 + 16 + lrow) * HD + k0 + bkoff);
        acc00 = WMMA_F16(a0, b0, acc00);
        acc10 = WMMA_F16(a1, b0, acc10);
        acc01 = WMMA_F16(a0, b1, acc01);
        acc11 = WMMA_F16(a1, b1, acc11);
    }
    int rbase = (lane < 16) ? 0 : 8;
#pragma unroll
    for (int r = 0; r < 8; ++r) {
        buh[(rbase + r) * BU_S + n0 + lrow]           = (_Float16)acc00[r];
        buh[(rbase + r) * BU_S + n0 + 16 + lrow]      = (_Float16)acc01[r];
        buh[(16 + rbase + r) * BU_S + n0 + lrow]      = (_Float16)acc10[r];
        buh[(16 + rbase + r) * BU_S + n0 + 16 + lrow] = (_Float16)acc11[r];
    }
}

// Pass 1: recompute Bu tile in LDS, local chunk scan, emit chunk-final 2-vectors.
__global__ __launch_bounds__(256)
void k_pass1(const float* __restrict__ x, const _Float16* __restrict__ BmatT,
             const float* __restrict__ prm, float* __restrict__ chunkfin) {
    __shared__ __align__(32) _Float16 xh[32 * XH_S];
    __shared__ __align__(32) _Float16 buh[32 * BU_S];
    int b = blockIdx.x / NCH, ch = blockIdx.x % NCH;
    int l0 = ch * CHUNK;
    const float* xb = x + ((size_t)b * SEQLEN + l0) * HD;
    for (int i = threadIdx.x; i < 32 * HD; i += 256) {
        int r = i >> 7, c = i & (HD - 1);
        xh[r * XH_S + c] = (_Float16)xb[(size_t)r * HD + c];
    }
    __syncthreads();
    gemm1_tile(xh, BmatT, buh, threadIdx.x);
    __syncthreads();
    int pt = threadIdx.x, p = pt >> 1;
    float m11 = PRM(0, p), m12 = PRM(1, p), m21 = PRM(2, p), m22 = PRM(3, p);
    float c1 = PRM(4, p), c2 = PRM(5, p);
    float s1 = 0.f, s2 = 0.f;
    for (int i = 0; i < CHUNK; ++i) {
        float u = (float)buh[i * BU_S + pt];
        float n1 = m11 * s1 + m12 * s2 + c1 * u;
        float n2 = m21 * s1 + m22 * s2 + c2 * u;
        s1 = n1; s2 = n2;
    }
    size_t o = (((size_t)b * NCH + ch) * PTD + pt) * 2;
    chunkfin[o] = s1; chunkfin[o + 1] = s2;
}

// Pass 2b: sequentially compose chunk finals into carry-in per chunk (via M^CHUNK).
__global__ __launch_bounds__(256)
void k_carries(const float* __restrict__ chunkfin, const float* __restrict__ prm,
               float* __restrict__ carries) {
    int tid = blockIdx.x * 256 + threadIdx.x;  // 0..2047
    int b = tid / PTD, pt = tid % PTD, p = pt >> 1;
    float a = PRM(6, p), bb = PRM(7, p), c = PRM(8, p), d = PRM(9, p);
    float c1 = 0.f, c2 = 0.f;
    for (int ch = 0; ch < NCH; ++ch) {
        size_t o = (((size_t)b * NCH + ch) * PTD + pt) * 2;
        carries[o] = c1; carries[o + 1] = c2;
        float lf1 = chunkfin[o], lf2 = chunkfin[o + 1];
        float n1 = lf1 + a * c1 + bb * c2;
        float n2 = lf2 + c * c1 + d * c2;
        c1 = n1; c2 = n2;
    }
}

// Pass 3: recompute Bu tile, scan with carry-in -> ys (f16) -> WMMA GEMM2 -> + x*D.
__global__ __launch_bounds__(256)
void k_pass2(const float* __restrict__ x, const _Float16* __restrict__ BmatT,
             const _Float16* __restrict__ CmatT, const float* __restrict__ carries,
             const float* __restrict__ prm, const float* __restrict__ Dv,
             float* __restrict__ out) {
    __shared__ __align__(32) _Float16 xh[32 * XH_S];
    __shared__ __align__(32) float    xf[32 * XF_S];
    __shared__ __align__(32) _Float16 buh[32 * BU_S];
    __shared__ __align__(32) _Float16 ysh[32 * BU_S];
    int b = blockIdx.x / NCH, ch = blockIdx.x % NCH;
    int l0 = ch * CHUNK;
    const float* xb = x + ((size_t)b * SEQLEN + l0) * HD;
    for (int i = threadIdx.x; i < 32 * HD; i += 256) {
        int r = i >> 7, c = i & (HD - 1);
        float v = xb[(size_t)r * HD + c];
        xf[r * XF_S + c] = v;
        xh[r * XH_S + c] = (_Float16)v;
    }
    __syncthreads();
    gemm1_tile(xh, BmatT, buh, threadIdx.x);
    __syncthreads();
    {
        int pt = threadIdx.x, p = pt >> 1;
        float m11 = PRM(0, p), m12 = PRM(1, p), m21 = PRM(2, p), m22 = PRM(3, p);
        float c1 = PRM(4, p), c2 = PRM(5, p);
        size_t co = (((size_t)b * NCH + ch) * PTD + pt) * 2;
        float s1 = carries[co], s2 = carries[co + 1];
        for (int i = 0; i < CHUNK; ++i) {
            float u = (float)buh[i * BU_S + pt];
            float n1 = m11 * s1 + m12 * s2 + c1 * u;
            float n2 = m21 * s1 + m22 * s2 + c2 * u;
            s1 = n1; s2 = n2;
            ysh[i * BU_S + pt] = (_Float16)s2;
        }
    }
    __syncthreads();
    // GEMM2: out32x128 = ysh(32x256) @ CmatT; wave 'w' owns h columns [16w,16w+16),
    // register-blocked over mt so the Cmat B-frag is loaded once per k-step.
    int wave = threadIdx.x >> 5, lane = threadIdx.x & 31;
    int lrow  = lane & 15;
    int khalf = (lane >> 4) * 8;
    int bkoff = (lane >> 4) * 16;
    int n0 = wave * 16;
    v8f acc0 = {}, acc1 = {};
#pragma unroll
    for (int k0 = 0; k0 < PTD; k0 += 32) {
        v16h bf = *(const v16h*)(CmatT + (n0 + lrow) * PTD + k0 + bkoff);
        v16h a0 = load_afrag(ysh + lrow * BU_S, k0, khalf);
        v16h a1 = load_afrag(ysh + (16 + lrow) * BU_S, k0, khalf);
        acc0 = WMMA_F16(a0, bf, acc0);
        acc1 = WMMA_F16(a1, bf, acc1);
    }
    int rbase = (lane < 16) ? 0 : 8;
    int hcol = n0 + lrow;
    float dv = Dv[hcol];
#pragma unroll
    for (int r = 0; r < 8; ++r) {
        int row0 = rbase + r;
        int row1 = 16 + rbase + r;
        size_t g0 = ((size_t)b * SEQLEN + l0 + row0) * HD + hcol;
        size_t g1 = ((size_t)b * SEQLEN + l0 + row1) * HD + hcol;
        out[g0] = acc0[r] + xf[row0 * XF_S + hcol] * dv;
        out[g1] = acc1[r] + xf[row1 * XF_S + hcol] * dv;
    }
}

extern "C" void kernel_launch(void* const* d_in, const int* in_sizes, int n_in,
                              void* d_out, int out_size, void* d_ws, size_t ws_size,
                              hipStream_t stream) {
    const float* x      = (const float*)d_in[0];
    const float* stepsp = (const float*)d_in[1];
    const float* Adiag  = (const float*)d_in[2];
    const float* Bin    = (const float*)d_in[3];
    const float* Cin    = (const float*)d_in[4];
    const float* Dv     = (const float*)d_in[5];
    float* out = (float*)d_out;
    float* ws  = (float*)d_ws;

    float*     chunkfin = ws + OFF_CF;
    float*     carries  = ws + OFF_CA;
    _Float16*  BmatT    = (_Float16*)(ws + OFF_BM);
    _Float16*  CmatT    = (_Float16*)(ws + OFF_CM);
    float*     prm      = ws + OFF_PR;

    k_params<<<1, 128, 0, stream>>>(stepsp, Adiag, Bin, Cin, BmatT, CmatT, prm);
    k_pass1<<<BATCHN * NCH, 256, 0, stream>>>(x, BmatT, prm, chunkfin);
    k_carries<<<(BATCHN * PTD) / 256, 256, 0, stream>>>(chunkfin, prm, carries);
    k_pass2<<<BATCHN * NCH, 256, 0, stream>>>(x, BmatT, CmatT, carries, prm, Dv, out);
}
